// Tconv_block_44976897524620
// MI455X (gfx1250) — matrix operand order repaired and checked by reading, not compiled, not run on hardware
//
#include <hip/hip_runtime.h>

// MI455X / gfx1250 masked transposed-conv upsampling block.
// Strategy: collapse low branch (1x1 then ConvT) into an effective 64->64
// ConvT; run both branches as one combined 128-out-channel implicit GEMM in
// bf16 WMMA (v_wmma_f32_16x16x32_bf16), phase-decomposed by output parity.

typedef __attribute__((ext_vector_type(16))) __bf16 v16bf;
typedef __attribute__((ext_vector_type(8)))  float  v8f;

struct Q2 { uint4 a, b; };
__device__ __forceinline__ v16bf frag_cast(uint4 a, uint4 b) {
  Q2 q; q.a = a; q.b = b;
  return __builtin_bit_cast(v16bf, q);
}

__device__ __forceinline__ unsigned short f2bf_rne(float f) {
  unsigned int u = __builtin_bit_cast(unsigned int, f);
  u += 0x7fffu + ((u >> 16) & 1u);         // round to nearest even
  return (unsigned short)(u >> 16);
}

// ---------------------------------------------------------------------------
// Prep 1: build all WMMA A-fragments (bf16, exact per-lane register layout).
// Frag F = ((path*9 + tap)*2 + ks)*4 + ct ; 144 frags x 32 lanes x 8 dwords.
// path0 = high weights as-is; path1 = low2_w contracted with low1_w over r.
// A-matrix 16x32 bf16 layout as packed here (same k order as B): M = lane&15,
// K(i,lane) = i + ((lane&16)?16:0), i = element index in v16bf.
// ---------------------------------------------------------------------------
__global__ void tb_prep_weights(const float* __restrict__ high_w,
                                const float* __restrict__ low1_w,
                                const float* __restrict__ low2_w,
                                unsigned int* __restrict__ wprep) {
  const int F    = blockIdx.x;      // 0..143
  const int lane = threadIdx.x;     // 0..31
  const int ct   = F & 3;
  const int ks   = (F >> 2) & 1;
  const int t3   = F >> 3;
  const int tap  = t3 % 9;
  const int path = t3 / 9;
  const int ky = tap / 3, kx = tap % 3;
  const int o   = ct * 16 + (lane & 15);
  const int hi8 = (lane & 16) ? 8 : 0;
  for (int idx = 0; idx < 8; ++idx) {
    int kbase = ks * 32 + 2 * hi8 + 2 * idx;
    float w0, w1;
    if (path == 0) {
      w0 = high_w[(( kbase      * 64 + o) * 3 + ky) * 3 + kx];
      w1 = high_w[(((kbase + 1) * 64 + o) * 3 + ky) * 3 + kx];
    } else {
      float s0 = 0.f, s1 = 0.f;
      for (int rr = 0; rr < 16; ++rr) {
        float w2 = low2_w[((rr * 64 + o) * 3 + ky) * 3 + kx];
        s0 += low1_w[rr * 64 + kbase]     * w2;
        s1 += low1_w[rr * 64 + kbase + 1] * w2;
      }
      w0 = s0; w1 = s1;
    }
    unsigned int p = (unsigned int)f2bf_rne(w0) |
                     ((unsigned int)f2bf_rne(w1) << 16);
    wprep[F * 256 + lane * 8 + idx] = p;
  }
}

// Prep 2: low1_b propagated through low2 conv, per tap (added only where the
// tap is inside the image, handled in the epilogue).
__global__ void tb_prep_blow(const float* __restrict__ low1_b,
                             const float* __restrict__ low2_w,
                             float* __restrict__ blow) {
  const int tap = blockIdx.x;   // 0..8
  const int o   = threadIdx.x;  // 0..63
  const int ky = tap / 3, kx = tap % 3;
  float s = 0.f;
  for (int rr = 0; rr < 16; ++rr)
    s += low1_b[rr] * low2_w[((rr * 64 + o) * 3 + ky) * 3 + kx];
  blow[tap * 64 + o] = s;
}

// ---------------------------------------------------------------------------
// Main: one WG = 1 batch x 1 output row-pair (2r,2r+1) x 64 output columns.
// LDS holds the bf16 input tile [2 rows][33(+1) cols][64 ch, pitch 72].
// B-fragment (32x16 bf16): N = lane&15, K = i + ((lane&16)?16:0)  -> the 16
// bf16 per lane are 32 contiguous bytes in LDS => two ds_load_b128.
// ---------------------------------------------------------------------------
#define LP 72  // channel pitch (ushorts) per (row,col): 144B, 16B aligned

__global__ __launch_bounds__(256) void tb_main(
    const float* __restrict__ x, const float* __restrict__ mask,
    const float* __restrict__ high_b, const float* __restrict__ low2_b,
    const unsigned int* __restrict__ wprep, const float* __restrict__ blow,
    float* __restrict__ out) {
  __shared__ __align__(16) unsigned short xs[2 * 34 * LP];

  const int b     = blockIdx.z;
  const int r     = blockIdx.y;   // input row; output rows 2r, 2r+1
  const int chunk = blockIdx.x;   // 64 output cols = input cols [32c,32c+32]
  const int c0    = chunk * 32;

  // Stage input tile as bf16 (zero-fill right/bottom halo).
  for (int e = threadIdx.x; e < 2 * 33 * 64; e += 256) {
    int lc = e % 33;
    int t  = e / 33;
    int k  = t & 63;
    int rowsel = t >> 6;
    int iy = r + rowsel;
    int ix = c0 + lc;
    float v = 0.f;
    if (iy < 256 && ix < 256)
      v = x[(((size_t)b * 64 + k) * 256 + iy) * 256 + ix];
    xs[(rowsel * 34 + lc) * LP + k] = f2bf_rne(v);
  }
  __syncthreads();

  const int lane = threadIdx.x & 31;
  const int wid  = threadIdx.x >> 5;
  const int ct   = wid & 3;        // output-channel tile (16 ch)
  const int nh   = wid >> 2;       // which half of the 8 pixel-tiles
  const int j    = lane & 15;
  const int hi16 = (lane & 16) ? 16 : 0;
  const int rowoff = (lane & 16) ? 8 : 0;
  const uint4* wq = (const uint4*)wprep;

  for (int jj = 0; jj < 4; ++jj) {
    // tap-count balanced job map: nh0 -> {0,3,5,6}, nh1 -> {1,2,4,7}
    int qi = nh * 4 + jj;
    int q  = (int)((0x74216530u >> (qi * 4)) & 7u);
    int ry = q >> 2, rx = (q >> 1) & 1, s = q & 1;

    v8f acch = {0, 0, 0, 0, 0, 0, 0, 0};
    v8f accl = {0, 0, 0, 0, 0, 0, 0, 0};

    auto tapstep = [&](int ky, int kx) {
      int rowsel  = (ky == 0) ? 1 : 0;                 // ky==0 reads row r+1
      int lc      = s * 16 + j + ((kx == 0) ? 1 : 0);  // kx==0 reads col+1
      int tap     = ky * 3 + kx;
      int busbase = (rowsel * 34 + lc) * LP + hi16;
#pragma unroll
      for (int ks = 0; ks < 2; ++ks) {
        const uint4* bp = (const uint4*)&xs[busbase + ks * 32];
        v16bf bf = frag_cast(bp[0], bp[1]);            // shared by both paths
        int Fh = ((0 * 9 + tap) * 2 + ks) * 4 + ct;
        int Fl = ((1 * 9 + tap) * 2 + ks) * 4 + ct;
        v16bf ah = frag_cast(wq[Fh * 64 + lane * 2], wq[Fh * 64 + lane * 2 + 1]);
        v16bf al = frag_cast(wq[Fl * 64 + lane * 2], wq[Fl * 64 + lane * 2 + 1]);
        acch = __builtin_amdgcn_wmma_f32_16x16x32_bf16(
            false, ah, false, bf, (short)0, acch, false, false);
        accl = __builtin_amdgcn_wmma_f32_16x16x32_bf16(
            false, al, false, bf, (short)0, accl, false, false);
      }
    };

    // phase (ry,rx): taps with oy = 2*iy - 1 + ky, ox = 2*ix - 1 + kx
    if (ry == 0) {
      if (rx == 0) { tapstep(1, 1); }
      else         { tapstep(1, 0); tapstep(1, 2); }
    } else {
      if (rx == 0) { tapstep(0, 1); tapstep(2, 1); }
      else         { tapstep(0, 0); tapstep(0, 2); tapstep(2, 0); tapstep(2, 2); }
    }

    // Epilogue: D layout 16x16 f32 -> pixel N = lane&15, row M = v + rowoff.
    int oy = 2 * r + ry;
    int ox = chunk * 64 + 2 * (s * 16 + j) + rx;
    float m  = mask[((size_t)b * 512 + oy) * 512 + ox];
    float im = 1.f - m;
    bool vy0 = (oy < 511);  // ky==0 tap needs input row r+1 < 256
    bool vx0 = (ox < 511);  // kx==0 tap needs input col +1  < 256
#pragma unroll
    for (int v = 0; v < 8; ++v) {
      int o = ct * 16 + v + rowoff;
      float bl = low2_b[o];
      if (ry == 0) {
        if (rx == 0) { bl += blow[4 * 64 + o]; }                 // (1,1)
        else { if (vx0) bl += blow[3 * 64 + o];                  // (1,0)
               bl += blow[5 * 64 + o]; }                         // (1,2)
      } else {
        if (rx == 0) { if (vy0) bl += blow[1 * 64 + o];          // (0,1)
                       bl += blow[7 * 64 + o]; }                 // (2,1)
        else {
          if (vy0 && vx0) bl += blow[0 * 64 + o];                // (0,0)
          if (vy0)        bl += blow[2 * 64 + o];                // (0,2)
          if (vx0)        bl += blow[6 * 64 + o];                // (2,0)
          bl += blow[8 * 64 + o];                                // (2,2)
        }
      }
      float res = m * (acch[v] + high_b[o]) + im * (accl[v] + bl);
      out[(((size_t)b * 64 + o) * 512 + oy) * 512 + ox] = res;
    }
  }
}

extern "C" void kernel_launch(void* const* d_in, const int* in_sizes, int n_in,
                              void* d_out, int out_size, void* d_ws, size_t ws_size,
                              hipStream_t stream) {
  const float* x      = (const float*)d_in[0];
  const float* mask   = (const float*)d_in[1];
  // d_in[2] inv_mask unused (identically 1 - mask; saves 4 MB of traffic)
  const float* high_w = (const float*)d_in[3];
  const float* high_b = (const float*)d_in[4];
  const float* low1_w = (const float*)d_in[5];
  const float* low1_b = (const float*)d_in[6];
  const float* low2_w = (const float*)d_in[7];
  const float* low2_b = (const float*)d_in[8];
  float* out = (float*)d_out;

  unsigned int* wprep = (unsigned int*)d_ws;                 // 147456 B
  float* blow = (float*)((char*)d_ws + 144 * 256 * 4);       // 2304 B

  tb_prep_weights<<<144, 32, 0, stream>>>(high_w, low1_w, low2_w, wprep);
  tb_prep_blow<<<9, 64, 0, stream>>>(low1_b, low2_w, blow);
  tb_main<<<dim3(8, 256, 4), 256, 0, stream>>>(x, mask, high_b, low2_b,
                                               wprep, blow, out);
}
